// EnTransformer_79413945303878
// MI455X (gfx1250) — compile-verified, matrix operand order checked by reading, not run on hardware
//
#include <hip/hip_runtime.h>
#include <math.h>

// ---------------- constants ----------------
#define NNODES 512
#define NVEC   8
#define NINV   128
#define NHEADS 8
#define DHEAD  16
#define NBESSEL 8
#define OUTINV 64
#define OUTVEC 4

typedef __attribute__((ext_vector_type(16))) __bf16 v16bf;
typedef __attribute__((ext_vector_type(8)))  float  v8f;
typedef __attribute__((ext_vector_type(4)))  unsigned int v4u;
typedef __attribute__((ext_vector_type(8)))  int    v8i;
typedef __attribute__((ext_vector_type(4)))  int    v4i;

// ---------------- small helpers ----------------
__device__ __forceinline__ unsigned short f2bf(float f) {
  union { float f; unsigned u; } v; v.f = f;
  unsigned r = v.u + 0x7fffu + ((v.u >> 16) & 1u);   // round-to-nearest-even
  return (unsigned short)(r >> 16);
}
__device__ __forceinline__ float bf2f(unsigned short u) {
  union { unsigned u; float f; } v; v.u = ((unsigned)u) << 16; return v.f;
}
__device__ __forceinline__ float silu(float x) { return x / (1.f + __expf(-x)); }

// WMMA bf16: A 16x32 tile (row llo, K-chunk by lane-hi half), B [N][K] layout.
__device__ __forceinline__ v8f wmma_bf16(const unsigned short* aPtr,
                                         const unsigned short* bPtr, v8f c) {
  v16bf a = *reinterpret_cast<const v16bf*>(aPtr);
  v16bf b = *reinterpret_cast<const v16bf*>(bPtr);
  return __builtin_amdgcn_wmma_f32_16x16x32_bf16(false, a, false, b, (short)0, c,
                                                 false, false);
}

// TDM: 1-D contiguous tensor load (nelem4 dwords, nelem4 < 65536) global -> LDS.
// Descriptor bit layout per CDNA5 ISA (#8.3 group0, #8.4 group1).
// This toolchain exposes the 6-arg builtin:
//   (uint32x4 g0, int32x8 g1, int32x4 g2, int32x4 g3, int32x8 extra, i32 cpol)
__device__ __forceinline__ void tdm_load_1d(const void* gaddr, unsigned ldsOff,
                                            unsigned nelem4) {
  unsigned long long ga = (unsigned long long)(size_t)gaddr;
  v4u g0;
  g0.x = 1u;                                        // count=1 (valid), user mode
  g0.y = ldsOff;                                    // lds_addr (bytes)
  g0.z = (unsigned)(ga & 0xffffffffull);            // global_addr[31:0]
  g0.w = (unsigned)((ga >> 32) & 0x1ffffffull) |    // global_addr[56:32]
         (2u << 30);                                // type=2 ("image")
  v8i g1;
  g1[0] = (int)(2u << 16);                          // wg_mask=0, data_size=2 (4B)
  g1[1] = (int)((nelem4 & 0xffffu) << 16);          // tensor_dim0[15:0] @ bits63:48
  g1[2] = (int)(((nelem4 >> 16) & 0xffffu) |        // tensor_dim0[31:16]
                (1u << 16));                        // tensor_dim1 = 1
  g1[3] = (int)(nelem4 << 16);                      // tile_dim0 @ bits127:112
  g1[4] = 0;                                        // tile_dim1=0, tile_dim2=0
  g1[5] = (int)nelem4;                              // tensor_dim0_stride[31:0]
  g1[6] = 0;
  g1[7] = 0;
  v4i z4 = {0, 0, 0, 0};
  v8i z8 = {0, 0, 0, 0, 0, 0, 0, 0};
  __builtin_amdgcn_tensor_load_to_lds(g0, g1, z4, z4, z8, 0);
}

__device__ __forceinline__ unsigned lds_offset_of(const void* p) {
  // generic LDS address: low 32 bits are the LDS byte offset
  return (unsigned)(unsigned long long)(size_t)p;
}

// ---------------- weight prep: [K][N] f32 -> [N][K] bf16 ----------------
__global__ void enb_prep_bf16_t(const float* __restrict__ src,
                                unsigned short* __restrict__ dst, int K, int N) {
  int idx = blockIdx.x * blockDim.x + threadIdx.x;
  if (idx >= K * N) return;
  int n = idx / K, k = idx % K;
  dst[idx] = f2bf(src[k * N + n]);
}

// ---------------- init: h = 1, pos = broadcast(x) ----------------
__global__ void enb_init(const float* __restrict__ x, float* __restrict__ h,
                         float* __restrict__ pos) {
  int idx = blockIdx.x * blockDim.x + threadIdx.x;
  if (idx < NNODES * NINV) {
    h[idx] = 1.f;
  } else {
    int j = idx - NNODES * NINV;
    if (j < NNODES * NVEC * 3) {
      int nd = j / (NVEC * 3);
      int k  = j % 3;
      pos[j] = x[nd * 3 + k];
    }
  }
}

// ---------------- per-node q / Kh / Vh projections (WMMA) ----------------
__global__ __launch_bounds__(128) void enb_node_pre(
    const float* __restrict__ h,
    const unsigned short* __restrict__ Wqt,
    const unsigned short* __restrict__ Wkht,
    const unsigned short* __restrict__ Wvht,
    float* __restrict__ qn, float* __restrict__ Kh, float* __restrict__ Vh) {
  __shared__ __align__(32) unsigned short hb[16][NINV];
  const int tid = threadIdx.x;
  const int m0  = blockIdx.x * 16;
  const int lane = tid & 31, wave = tid >> 5;
  const int lhi = lane >> 4, llo = lane & 15;

#pragma unroll
  for (int e = 0; e < 16; ++e) hb[e][tid] = f2bf(h[(m0 + e) * NINV + tid]);
  __syncthreads();

  const unsigned short* Ws[3] = {Wqt, Wkht, Wvht};
  float* Os[3] = {qn, Kh, Vh};
#pragma unroll
  for (int mat = 0; mat < 3; ++mat) {
#pragma unroll
    for (int i = 0; i < 2; ++i) {
      int ct = wave * 2 + i;
      v8f acc = {};
#pragma unroll
      for (int ks = 0; ks < 4; ++ks)
        acc = wmma_bf16(&hb[llo][ks * 32 + lhi * 16],
                        &Ws[mat][(ct * 16 + llo) * NINV + ks * 32 + lhi * 16], acc);
#pragma unroll
      for (int v = 0; v < 8; ++v) {
        int m = v + 8 * lhi;
        Os[mat][(m0 + m) * NINV + ct * 16 + llo] = acc[v];
      }
    }
  }
}

// ---------------- fused per-receiver attention block ----------------
__global__ __launch_bounds__(128) void enb_block(
    const float* __restrict__ posIn, float* __restrict__ posOut,
    float* __restrict__ h,
    const float* __restrict__ qn, const float* __restrict__ Kh,
    const float* __restrict__ Vh,
    const unsigned short* __restrict__ Wem0t,  // [128][64]  bf16
    const unsigned short* __restrict__ Wem1t,  // [128][128] bf16
    const unsigned short* __restrict__ Wket,   // [128][128] bf16
    const unsigned short* __restrict__ Wvet,   // [128][128] bf16
    const float* __restrict__ Wgate,           // [128][8]   f32
    const float* __restrict__ oW0, const float* __restrict__ ob0,
    const float* __restrict__ oW1, const float* __restrict__ ob1,
    const float* __restrict__ oW2, const float* __restrict__ ob2) {
  const int r   = blockIdx.x;
  const int tid = threadIdx.x;
  const int lane = tid & 31, wave = tid >> 5;
  const int lhi = lane >> 4, llo = lane & 15;

  __shared__ __align__(32) unsigned short X0[16][64];
  __shared__ __align__(32) unsigned short X1[16][NINV];
  __shared__ __align__(32) unsigned short Ebf[16][NINV];
  __shared__ __align__(16) float KV[2][16][NINV];  // TDM-staged Kh/Vh -> k/v
  __shared__ float Ush[16][NVEC][3];
  __shared__ float Gsh[16][NVEC];
  __shared__ float Lsh[16][NHEADS];
  __shared__ float Wsh[16][NHEADS];
  __shared__ float scaleL[NHEADS], mrun[NHEADS], ssum[NHEADS];
  __shared__ float qr[NINV], posR[NVEC][3];
  __shared__ float ha[NINV], t1b[NINV], t2b[NINV];

  qr[tid] = qn[r * NINV + tid];
  if (tid < NVEC * 3) posR[tid / 3][tid % 3] = posIn[r * NVEC * 3 + tid];
  if (tid < NHEADS) { mrun[tid] = -1e30f; ssum[tid] = 0.f; }
  float hacc = 0.f, vacc = 0.f;
  const unsigned kvOff0 = lds_offset_of(&KV[0][0][0]);
  const unsigned kvOff1 = lds_offset_of(&KV[1][0][0]);
  __syncthreads();

  for (int t = 0; t < NNODES / 16; ++t) {
    // ---- TDM: async-stage Kh/Vh 16x128 tiles (contiguous 8KB each) ----
    if (wave == 0) {
      tdm_load_1d(Kh + (size_t)t * 16 * NINV, kvOff0, 16 * NINV);
      tdm_load_1d(Vh + (size_t)t * 16 * NINV, kvOff1, 16 * NINV);
    }

    // ---- geometry + radial embedding (overlaps the DMA) ----
    {
      const int e = tid >> 3, vc = tid & 7;
      const int s = t * 16 + e;
      float vx = posIn[(s * NVEC + vc) * 3 + 0] - posR[vc][0];
      float vy = posIn[(s * NVEC + vc) * 3 + 1] - posR[vc][1];
      float vz = posIn[(s * NVEC + vc) * 3 + 2] - posR[vc][2];
      float len  = sqrtf(vx * vx + vy * vy + vz * vz + 1e-12f);
      float rinv = 1.f / len;
      Ush[e][vc][0] = vx * rinv;
      Ush[e][vc][1] = vy * rinv;
      Ush[e][vc][2] = vz * rinv;
      float tt  = 2.f * (1.f - len * 0.1f);
      float env = (tt > 0.f) ? 1.9784649f * __expf(-1.f / tt) : 0.f;  // 1.2*e^0.5
      float cb  = 0.4472136f * rinv * env;                            // sqrt(2/10)/r*env
#pragma unroll
      for (int bi = 0; bi < NBESSEL; ++bi)
        X0[e][vc * NBESSEL + bi] =
            f2bf(cb * __sinf(len * (float)(bi + 1) * 0.31415927f));
    }
    __syncthreads();

    // ---- GEMM1: X0(16x64) @ Wem0 -> silu -> X1 (bf16) ----
#pragma unroll
    for (int i = 0; i < 2; ++i) {
      int ct = wave * 2 + i;
      v8f acc = {};
#pragma unroll
      for (int ks = 0; ks < 2; ++ks)
        acc = wmma_bf16(&X0[llo][ks * 32 + lhi * 16],
                        &Wem0t[(ct * 16 + llo) * 64 + ks * 32 + lhi * 16], acc);
#pragma unroll
      for (int v = 0; v < 8; ++v)
        X1[v + 8 * lhi][ct * 16 + llo] = f2bf(silu(acc[v]));
    }
    __syncthreads();

    // ---- GEMM2: X1(16x128) @ Wem1 -> silu -> Ebf ----
#pragma unroll
    for (int i = 0; i < 2; ++i) {
      int ct = wave * 2 + i;
      v8f acc = {};
#pragma unroll
      for (int ks = 0; ks < 4; ++ks)
        acc = wmma_bf16(&X1[llo][ks * 32 + lhi * 16],
                        &Wem1t[(ct * 16 + llo) * NINV + ks * 32 + lhi * 16], acc);
#pragma unroll
      for (int v = 0; v < 8; ++v)
        Ebf[v + 8 * lhi][ct * 16 + llo] = f2bf(silu(acc[v]));
    }
    // drain the tensor DMA before anyone touches KV
    if (wave == 0) __builtin_amdgcn_s_wait_tensorcnt(0);
    __syncthreads();

    // ---- GEMM3: k = (E@Wk_e)*Kh , v = (E@Wv_e)*Vh ; plus gate ----
#pragma unroll
    for (int mat = 0; mat < 2; ++mat) {
      const unsigned short* Wt = mat ? Wvet : Wket;
#pragma unroll
      for (int i = 0; i < 2; ++i) {
        int ct = wave * 2 + i;
        v8f acc = {};
#pragma unroll
        for (int ks = 0; ks < 4; ++ks)
          acc = wmma_bf16(&Ebf[llo][ks * 32 + lhi * 16],
                          &Wt[(ct * 16 + llo) * NINV + ks * 32 + lhi * 16], acc);
#pragma unroll
        for (int v = 0; v < 8; ++v) {
          int m = v + 8 * lhi, n = ct * 16 + llo;
          KV[mat][m][n] = acc[v] * KV[mat][m][n];
        }
      }
    }
    {  // gate = E @ Wgate  (128->8, VALU)
      const int e = tid >> 3, g = tid & 7;
      float s = 0.f;
#pragma unroll
      for (int k = 0; k < NINV; ++k) s += bf2f(Ebf[e][k]) * Wgate[k * NVEC + g];
      Gsh[e][g] = s;
    }
    __syncthreads();

    // ---- logits ----
    {
      const int e = tid >> 3, hd = tid & 7;
      float l = 0.f;
#pragma unroll
      for (int d = 0; d < DHEAD; ++d) l += qr[hd * DHEAD + d] * KV[0][e][hd * DHEAD + d];
      l *= 0.25f;                       // 1/sqrt(D_HEAD)
      if (t * 16 + e == r) l = -1e30f;  // mask self-edge
      Lsh[e][hd] = l;
    }
    __syncthreads();

    // ---- online softmax update (one thread per head) ----
    if (tid < NHEADS) {
      float tm = -1e30f;
#pragma unroll
      for (int e = 0; e < 16; ++e) tm = fmaxf(tm, Lsh[e][tid]);
      float mn = fmaxf(mrun[tid], tm);
      float sc = __expf(mrun[tid] - mn);
      float add = 0.f;
#pragma unroll
      for (int e = 0; e < 16; ++e) {
        float w = __expf(Lsh[e][tid] - mn);
        Wsh[e][tid] = w;
        add += w;
      }
      ssum[tid]   = ssum[tid] * sc + add;
      mrun[tid]   = mn;
      scaleL[tid] = sc;
    }
    __syncthreads();

    // ---- accumulate h_agg / vec_agg ----
    {
      float sc = scaleL[tid >> 4];
      float a = 0.f;
#pragma unroll
      for (int e = 0; e < 16; ++e) a += Wsh[e][tid >> 4] * KV[1][e][tid];
      hacc = hacc * sc + a;
    }
    if (tid < NVEC * 3) {
      int vc = tid / 3, k = tid % 3;
      float sc = scaleL[vc];
      float a = 0.f;
#pragma unroll
      for (int e = 0; e < 16; ++e)
        a += Wsh[e][vc] * Gsh[e][vc] * Ush[e][vc][k];
      vacc = vacc * sc + a * 1.7320508f;  // sqrt(3) from l=1 SH
    }
    __syncthreads();
  }

  // ---- finalize attention ----
  ha[tid] = hacc / ssum[tid >> 4];
  if (tid < NVEC * 3) {
    int vc = tid / 3, k = tid % 3;
    posOut[(r * NVEC + vc) * 3 + k] = posR[vc][k] + vacc / ssum[vc];
  }
  __syncthreads();

  // ---- out MLP (silu, silu, linear) + residual on h ----
  {
    float z = ob0[tid];
#pragma unroll
    for (int k = 0; k < NINV; ++k) z += ha[k] * oW0[k * NINV + tid];
    t1b[tid] = silu(z);
  }
  __syncthreads();
  {
    float z = ob1[tid];
#pragma unroll
    for (int k = 0; k < NINV; ++k) z += t1b[k] * oW1[k * NINV + tid];
    t2b[tid] = silu(z);
  }
  __syncthreads();
  {
    float z = ob2[tid];
#pragma unroll
    for (int k = 0; k < NINV; ++k) z += t2b[k] * oW2[k * NINV + tid];
    h[r * NINV + tid] += z;
  }
}

// ---------------- readout ----------------
__global__ __launch_bounds__(64) void enb_readout(
    const float* __restrict__ pos, const float* __restrict__ h,
    const float* __restrict__ Wvec, const float* __restrict__ Winv,
    const float* __restrict__ Wfin, const float* __restrict__ bfin,
    float* __restrict__ out) {
  const int r = blockIdx.x, tid = threadIdx.x;
  __shared__ float P[NVEC][3], com[3], e0[OUTINV];
  if (tid < NVEC * 3) P[tid / 3][tid % 3] = pos[r * NVEC * 3 + tid];
  __syncthreads();
  if (tid < 3) {
    float s = 0.f;
#pragma unroll
    for (int i = 0; i < NVEC; ++i) s += P[i][tid];
    com[tid] = s * 0.125f;
  }
  __syncthreads();
  if (tid < OUTVEC * 3) {
    int o = tid / 3, k = tid % 3;
    float s = 0.f;
#pragma unroll
    for (int i = 0; i < NVEC; ++i) s += (P[i][k] - com[k]) * Wvec[i * OUTVEC + o];
    out[r * OUTVEC * 3 + o * 3 + k] = com[k] + s * 0.35355339f;  // 1/sqrt(8)
  }
  {
    float z = 0.f;
#pragma unroll
    for (int k = 0; k < NINV; ++k) z += h[r * NINV + k] * Winv[k * OUTINV + tid];
    z *= 0.08838835f;                       // 1/sqrt(128)
    e0[tid] = (z > 0.f) ? z : __expf(z) - 1.f;  // elu
  }
  __syncthreads();
  {
    float z = bfin[tid];
#pragma unroll
    for (int j = 0; j < OUTINV; ++j) z += e0[j] * Wfin[j * OUTINV + tid];
    out[NNODES * OUTVEC * 3 + r * OUTINV + tid] = z;
  }
}

// ---------------- host ----------------
extern "C" void kernel_launch(void* const* d_in, const int* in_sizes, int n_in,
                              void* d_out, int out_size, void* d_ws, size_t ws_size,
                              hipStream_t stream) {
  (void)in_sizes; (void)n_in; (void)out_size; (void)ws_size;
  // Input order: x, then params leaves (sorted-key tree order):
  //  1 W_final, 2 W_inv, 3 W_vec, 4 b_final, then per block (base = 5 + 14*b):
  //  +0 Wgate +1 Wk_e +2 Wk_h +3 Wq +4 Wv_e +5 Wv_h +6 em0 +7 em1
  //  +8 oW0 +9 ob0 +10 oW1 +11 ob1 +12 oW2 +13 ob2
  const float* x       = (const float*)d_in[0];
  const float* W_final = (const float*)d_in[1];
  const float* W_inv   = (const float*)d_in[2];
  const float* W_vec   = (const float*)d_in[3];
  const float* b_final = (const float*)d_in[4];

  char* wsb = (char*)d_ws;
  size_t off = 0;
  auto alloc = [&](size_t bytes) -> void* {
    void* p = wsb + off;
    off += (bytes + 255) & ~(size_t)255;
    return p;
  };
  float* h    = (float*)alloc(NNODES * NINV * sizeof(float));
  float* posA = (float*)alloc(NNODES * NVEC * 3 * sizeof(float));
  float* posB = (float*)alloc(NNODES * NVEC * 3 * sizeof(float));
  float* qn   = (float*)alloc(NNODES * NINV * sizeof(float));
  float* KhB  = (float*)alloc(NNODES * NINV * sizeof(float));
  float* VhB  = (float*)alloc(NNODES * NINV * sizeof(float));
  // bf16 transposed weights per block: em0, em1, ke, ve, q, kh, vh
  unsigned short* Wt[2][7];
  for (int b = 0; b < 2; ++b) {
    Wt[b][0] = (unsigned short*)alloc(128 * 64 * sizeof(unsigned short));
    for (int m = 1; m < 7; ++m)
      Wt[b][m] = (unsigned short*)alloc(128 * 128 * sizeof(unsigned short));
  }

  // weight prep
  for (int b = 0; b < 2; ++b) {
    int base = 5 + 14 * b;
    const float* em0  = (const float*)d_in[base + 6];
    const float* em1  = (const float*)d_in[base + 7];
    const float* Wk_e = (const float*)d_in[base + 1];
    const float* Wv_e = (const float*)d_in[base + 4];
    const float* Wq   = (const float*)d_in[base + 3];
    const float* Wk_h = (const float*)d_in[base + 2];
    const float* Wv_h = (const float*)d_in[base + 5];
    enb_prep_bf16_t<<<(64 * 128 + 255) / 256, 256, 0, stream>>>(em0, Wt[b][0], 64, 128);
    enb_prep_bf16_t<<<(128 * 128 + 255) / 256, 256, 0, stream>>>(em1, Wt[b][1], 128, 128);
    enb_prep_bf16_t<<<(128 * 128 + 255) / 256, 256, 0, stream>>>(Wk_e, Wt[b][2], 128, 128);
    enb_prep_bf16_t<<<(128 * 128 + 255) / 256, 256, 0, stream>>>(Wv_e, Wt[b][3], 128, 128);
    enb_prep_bf16_t<<<(128 * 128 + 255) / 256, 256, 0, stream>>>(Wq,   Wt[b][4], 128, 128);
    enb_prep_bf16_t<<<(128 * 128 + 255) / 256, 256, 0, stream>>>(Wk_h, Wt[b][5], 128, 128);
    enb_prep_bf16_t<<<(128 * 128 + 255) / 256, 256, 0, stream>>>(Wv_h, Wt[b][6], 128, 128);
  }

  // init
  {
    int total = NNODES * NINV + NNODES * NVEC * 3;
    enb_init<<<(total + 255) / 256, 256, 0, stream>>>(x, h, posA);
  }

  const float* pin = posA;
  float* pout = posB;
  for (int b = 0; b < 2; ++b) {
    int base = 5 + 14 * b;
    enb_node_pre<<<NNODES / 16, 128, 0, stream>>>(h, Wt[b][4], Wt[b][5], Wt[b][6],
                                                  qn, KhB, VhB);
    enb_block<<<NNODES, 128, 0, stream>>>(
        pin, pout, h, qn, KhB, VhB,
        Wt[b][0], Wt[b][1], Wt[b][2], Wt[b][3],
        (const float*)d_in[base + 0],                        // Wgate
        (const float*)d_in[base + 8],  (const float*)d_in[base + 9],
        (const float*)d_in[base + 10], (const float*)d_in[base + 11],
        (const float*)d_in[base + 12], (const float*)d_in[base + 13]);
    const float* tmp = pin;
    pin = pout;
    pout = (float*)tmp;
  }

  enb_readout<<<NNODES, 64, 0, stream>>>(pin, h, W_vec, W_inv, W_final, b_final,
                                         (float*)d_out);
}
